// SE3Transformer_11398843204127
// MI455X (gfx1250) — matrix operand order repaired and checked by reading, not compiled, not run on hardware
//
#include <hip/hip_runtime.h>
#include <hip/hip_bf16.h>
#include <math.h>

#define EDGES 65536
#define NODES 8192
#define CDIV(a,b) (((a)+(b)-1)/(b))
#define EPS 1e-8f

typedef __attribute__((ext_vector_type(16))) _Float16 v16h;
typedef __attribute__((ext_vector_type(8)))  float    v8f;

// ---------------------------------------------------------------------------
// RCG (real Clebsch-Gordan) generation on device (compile-time consts in ref)
// ---------------------------------------------------------------------------
struct cpx { double re, im; };
__device__ inline cpx cmul(cpx a, cpx b){ cpx r; r.re=a.re*b.re-a.im*b.im; r.im=a.re*b.im+a.im*b.re; return r; }
__device__ inline cpx cconj(cpx a){ cpx r; r.re=a.re; r.im=-a.im; return r; }

__device__ inline double dfac(int n){
  const double f[10] = {1.,1.,2.,6.,24.,120.,720.,5040.,40320.,362880.};
  return f[n];
}

__device__ double cgcoef(int j1,int m1,int j2,int m2,int j3,int m3){
  if (m1+m2 != m3) return 0.0;
  if (j3 < abs(j1-j2) || j3 > j1+j2) return 0.0;
  double pre = sqrt((2.0*j3+1.0)*dfac(j3+j1-j2)*dfac(j3-j1+j2)*dfac(j1+j2-j3)/dfac(j1+j2+j3+1));
  pre *= sqrt(dfac(j3+m3)*dfac(j3-m3)*dfac(j1-m1)*dfac(j1+m1)*dfac(j2-m2)*dfac(j2+m2));
  double s = 0.0;
  for (int k=0; k<=j1+j2-j3; ++k){
    int d1=j1+j2-j3-k, d2=j1-m1-k, d3=j2+m2-k, d4=j3-j2+m1+k, d5=j3-j1-m2+k;
    if (d1<0||d2<0||d3<0||d4<0||d5<0) continue;
    double t = dfac(k)*dfac(d1)*dfac(d2)*dfac(d3)*dfac(d4)*dfac(d5);
    s += ((k&1)? -1.0:1.0)/t;
  }
  return pre*s;
}

__device__ void qmat(int l, cpx* Q, int ld){
  int dim = 2*l+1;
  for (int i=0;i<dim;++i) for (int j=0;j<dim;++j){ Q[i*ld+j].re=0.0; Q[i*ld+j].im=0.0; }
  const double is2 = 0.70710678118654752440;
  for (int m=-l;m<=l;++m){
    if (m>0){
      Q[(l+m)*ld+(l-m)].re = is2;
      Q[(l+m)*ld+(l+m)].re = (m&1)? -is2 : is2;
    } else if (m==0){
      Q[l*ld+l].re = 1.0;
    } else {
      int u = -m;
      Q[(l+m)*ld+(l+m)].im = is2;
      Q[(l+m)*ld+(l+u)].im = -(((u&1)? -1.0 : 1.0) * is2);
    }
  }
}

__device__ void compute_rcg(int li, int lo, int J, float* out){
  int od=2*lo+1, id=2*li+1, Jd=2*J+1;
  double Cc[5][5][9];
  for (int a=0;a<5;++a) for (int b=0;b<5;++b) for (int c=0;c<9;++c) Cc[a][b][c]=0.0;
  for (int Mo=-lo;Mo<=lo;++Mo)
    for (int Mi=-li;Mi<=li;++Mi){
      int MJ = Mo-Mi;
      if (MJ>=-J && MJ<=J) Cc[lo+Mo][li+Mi][J+MJ] = cgcoef(li,Mi,J,MJ,lo,Mo);
    }
  cpx Qo[25], Qi[25], QJ[81];
  qmat(lo,Qo,5); qmat(li,Qi,5); qmat(J,QJ,9);
  double Cre[5][5][9], Cim[5][5][9];
  double mre=0.0, mim=0.0;
  for (int a=0;a<od;++a)
    for (int e=0;e<id;++e)
      for (int f=0;f<Jd;++f){
        double sre=0.0, sim=0.0;
        for (int b=0;b<od;++b)
          for (int c=0;c<id;++c)
            for (int d=0;d<Jd;++d){
              double cc = Cc[b][c][d];
              if (cc==0.0) continue;
              cpx t = cmul(Qo[a*5+b], cconj(Qi[e*5+c]));
              t = cmul(t, cconj(QJ[f*9+d]));
              sre += t.re*cc; sim += t.im*cc;
            }
        Cre[a][e][f]=sre; Cim[a][e][f]=sim;
        mre = fmax(mre, fabs(sre)); mim = fmax(mim, fabs(sim));
      }
  int useRe = (mre >= mim);
  for (int a=0;a<od;++a)
    for (int e=0;e<id;++e)
      for (int f=0;f<Jd;++f)
        out[(a*id+e)*Jd+f] = (float)(useRe ? Cre[a][e][f] : Cim[a][e][f]);
}

__global__ void rcg_kernel(float* __restrict__ rcg){
  int want = threadIdx.x;
  int idx = 0, off = 0;
  for (int di=0; di<3; ++di)
    for (int dg=0; dg<3; ++dg){
      int Jmin = abs(di-dg), Jmax = di+dg;
      for (int J=Jmin; J<=Jmax; ++J){
        int sz = (2*dg+1)*(2*di+1)*(2*J+1);
        if (idx == want) compute_rcg(di, dg, J, rcg + off);
        idx++; off += sz;
      }
    }
}

// ---------------------------------------------------------------------------
// Edge geometry: rel, r, rhat, rfeat(f16, padded K=64), spherical harmonics
// ---------------------------------------------------------------------------
__global__ void edge_geom_kernel(const float* __restrict__ pos, const int* __restrict__ src,
                                 const int* __restrict__ dst, const float* __restrict__ efeat,
                                 _Float16* __restrict__ rf16, float* __restrict__ Yout)
{
  int e = blockIdx.x*blockDim.x + threadIdx.x;
  if (e >= EDGES) return;
  int s = src[e], d = dst[e];
  float rx = pos[3*d+0]-pos[3*s+0];
  float ry = pos[3*d+1]-pos[3*s+1];
  float rz = pos[3*d+2]-pos[3*s+2];
  float r = sqrtf(rx*rx+ry*ry+rz*rz + 1e-6f);
  float inv = 1.0f/r;
  float x = rx*inv, y = ry*inv, z = rz*inv;

  _Float16* rf = rf16 + (size_t)e*64;
  #pragma unroll
  for (int c=0;c<32;++c) rf[c] = (_Float16)efeat[(size_t)e*32+c];
  rf[32] = (_Float16)r;
  #pragma unroll
  for (int c=33;c<64;++c) rf[c] = (_Float16)0.0f;

  float ct = fminf(fmaxf(z, -1.0f+1e-6f), 1.0f-1e-6f);
  float st = sqrtf(1.0f-ct*ct);
  float phi = atan2f(y, x);

  float P[5][5];
  P[0][0]=1.0f;
  {
    const float df[5]={1.f,1.f,3.f,15.f,105.f};
    float stp=1.0f;
    for (int m=1;m<=4;++m){ stp*=st; P[m][m] = ((m&1)? -1.0f:1.0f)*df[m]*stp; }
  }
  for (int m=0;m<4;++m) P[m+1][m] = (2.0f*(float)m+1.0f)*ct*P[m][m];
  for (int m=0;m<=4;++m)
    for (int l=m+2;l<=4;++l)
      P[l][m] = ((2.0f*(float)l-1.0f)*ct*P[l-1][m] - (float)(l+m-1)*P[l-2][m])/(float)(l-m);

  float cm[5], sm[5];
  cm[0]=1.f; sm[0]=0.f;
  float c1=cosf(phi), s1=sinf(phi);
  for (int m=1;m<=4;++m){
    float cp = cm[m-1]*c1 - sm[m-1]*s1;
    float sp = sm[m-1]*c1 + cm[m-1]*s1;
    cm[m]=cp; sm[m]=sp;
  }
  const float FACT[9]={1,1,2,6,24,120,720,5040,40320};
  const float SQRT2 = 1.41421356237309f;
  float* Ye = Yout + (size_t)e*25;
  for (int l=0;l<=4;++l){
    for (int m=-l;m<=l;++m){
      int am = (m<0)? -m:m;
      float K = sqrtf((2.0f*(float)l+1.0f)/(4.0f*3.14159265358979f) * FACT[l-am]/FACT[l+am]);
      float v;
      if (m==0)      v = K*P[l][0];
      else if (m>0)  v = ((m&1)?-1.0f:1.0f)*SQRT2*K*cm[m]*P[l][m];
      else           v = ((am&1)?-1.0f:1.0f)*SQRT2*K*sm[am]*P[l][am];
      Ye[l*l + (m+l)] = v;
    }
  }
}

// ---------------------------------------------------------------------------
// Basis: basis[di,do][e,j,o,i] = sum_m RCG[di,do,J][o,i,m] * Y[J][e,m]
// ---------------------------------------------------------------------------
__global__ void basis_kernel(const float* __restrict__ Y, const float* __restrict__ rcg,
                             float* __restrict__ basis)
{
  int e = blockIdx.x*blockDim.x + threadIdx.x;
  if (e >= EDGES) return;
  const float* Ye = Y + (size_t)e*25;
  size_t pairStart = 0;
  int ro = 0;
  for (int di=0; di<3; ++di)
    for (int dg=0; dg<3; ++dg){
      int Jmin = abs(di-dg), Jmax = di+dg;
      int od = 2*dg+1, id = 2*di+1;
      int nJ = Jmax-Jmin+1;
      int sz = nJ*od*id;
      float* Be = basis + pairStart + (size_t)e*sz;
      for (int jj=0; jj<nJ; ++jj){
        int J = Jmin+jj, Jd = 2*J+1;
        const float* C = rcg + ro;
        const float* Yj = Ye + J*J;
        for (int o=0;o<od;++o)
          for (int i=0;i<id;++i){
            float s = 0.f;
            for (int m=0;m<Jd;++m) s += C[(o*id+i)*Jd+m]*Yj[m];
            Be[(jj*od+o)*id+i] = s;
          }
        ro += od*id*Jd;
      }
      pairStart += (size_t)EDGES*sz;
    }
}

// ---------------------------------------------------------------------------
// Weight packing into WMMA B-operand lane layout (f16)
// B 32x16 f16: lane -> N = lane&15; half h -> K = (h>>3)*16 + ((lane&16)?8:0) + (h&7)
// ---------------------------------------------------------------------------
__global__ __launch_bounds__(32)
void pack_mlp_kernel(const float* __restrict__ w0, const float* __restrict__ w1,
                     const float* __restrict__ w2, _Float16* __restrict__ dstp,
                     int N, int Npad)
{
  int lane = threadIdx.x;
  int t = blockIdx.x;
  const float* W; int K, ldN, kt, nt, Ncols;
  if (t < 4)      { W = w0; K = 33; ldN = 32; kt = t >> 1; nt = t & 1;  Ncols = 32; }
  else if (t < 6) { W = w1; K = 32; ldN = 32; kt = 0;      nt = t - 4;  Ncols = 32; }
  else            { W = w2; K = 32; ldN = N;  kt = 0;      nt = t - 6;  Ncols = N;  }
  (void)Npad;
  _Float16* o = dstp + (size_t)t*512 + lane*16;
  int n = nt*16 + (lane & 15);
  int kb = (lane & 16) ? 8 : 0;
  #pragma unroll
  for (int h=0; h<16; ++h){
    int kk = kt*32 + (h>>3)*16 + kb + (h&7);
    float v = (kk < K && n < Ncols) ? W[(size_t)kk*ldN + n] : 0.f;
    o[h] = (_Float16)v;
  }
}

// ---------------------------------------------------------------------------
// Radial-MLP batched GEMM via WMMA: per wave, 16 edges through 33->32->32->N
// A 16x32 f16 layout: lane -> M = lane&15; half h -> K = (h>>3)*16 + ((lane&16)?8:0) + (h&7)
// C/D 16x16 f32: lane -> N = lane&15; reg r -> M = r + ((lane&16)?8:0)
// ---------------------------------------------------------------------------
__global__ __launch_bounds__(32)
void mlp_wmma_kernel(const _Float16* __restrict__ X,   // EDGES x 64 (rfeat f16, K-padded)
                     const _Float16* __restrict__ wp,  // packed tiles: W0(4) W1(2) W2(Npad/16)
                     const float* __restrict__ b0,
                     const float* __restrict__ b1,
                     const float* __restrict__ b2,
                     float* __restrict__ out, int N, int Npad)
{
  __shared__ _Float16 ldsX[16*64];
  __shared__ _Float16 ldsH[16*32];
  int lane = threadIdx.x;
  int e0 = blockIdx.x * 16;

  // coalesced stage of the 16x64-half A tile into LDS
  {
    const uint32_t* s = (const uint32_t*)(X + (size_t)e0*64);
    uint32_t* d = (uint32_t*)ldsX;
    #pragma unroll
    for (int i=0;i<16;++i) d[lane + 32*i] = s[lane + 32*i];
  }
  if (e0 + 16 < EDGES)
    __builtin_prefetch(X + (size_t)(e0+16)*64, 0, 3);   // global_prefetch_b8
  __syncthreads();

  int Mrow = lane & 15;
  int hsel = (lane & 16) ? 8 : 0;

  // -------- stage 0: H0 = relu(X @ W0 + b0), K=64 (2 ksteps), N=32 (2 tiles)
  v8f acc[2];
  #pragma unroll
  for (int nt=0; nt<2; ++nt){
    v8f c = {};
    #pragma unroll
    for (int kt=0; kt<2; ++kt){
      v16h a;
      #pragma unroll
      for (int h=0; h<16; ++h){
        int kk = (h>>3)*16 + hsel + (h&7);
        a[h] = ldsX[Mrow*64 + kt*32 + kk];
      }
      v16h b = *(const v16h*)(wp + (size_t)(kt*2+nt)*512 + lane*16);
      c = __builtin_amdgcn_wmma_f32_16x16x32_f16(false, a, false, b, (short)0, c,
                                                 false, false);
    }
    acc[nt] = c;
  }
  #pragma unroll
  for (int nt=0; nt<2; ++nt){
    int n = nt*16 + (lane & 15);
    float bias = b0[n];
    #pragma unroll
    for (int r=0; r<8; ++r){
      int m = r + hsel;
      float v = acc[nt][r] + bias;
      ldsH[m*32 + n] = (_Float16)(v > 0.f ? v : 0.f);
    }
  }
  __syncthreads();

  // -------- stage 1: H1 = relu(H0 @ W1 + b1), K=32, N=32
  const _Float16* wp1 = wp + 4*512;
  v16h aH;
  #pragma unroll
  for (int h=0; h<16; ++h){
    int kk = (h>>3)*16 + hsel + (h&7);
    aH[h] = ldsH[Mrow*32 + kk];
  }
  #pragma unroll
  for (int nt=0; nt<2; ++nt){
    v16h b = *(const v16h*)(wp1 + (size_t)nt*512 + lane*16);
    v8f c = {};
    acc[nt] = __builtin_amdgcn_wmma_f32_16x16x32_f16(false, aH, false, b, (short)0, c,
                                                     false, false);
  }
  __syncthreads();
  #pragma unroll
  for (int nt=0; nt<2; ++nt){
    int n = nt*16 + (lane & 15);
    float bias = b1[n];
    #pragma unroll
    for (int r=0; r<8; ++r){
      int m = r + hsel;
      float v = acc[nt][r] + bias;
      ldsH[m*32 + n] = (_Float16)(v > 0.f ? v : 0.f);
    }
  }
  __syncthreads();

  // -------- stage 2: R = H1 @ W2 + b2, K=32, N tiles = Npad/16
  const _Float16* wp2 = wp + 6*512;
  #pragma unroll
  for (int h=0; h<16; ++h){
    int kk = (h>>3)*16 + hsel + (h&7);
    aH[h] = ldsH[Mrow*32 + kk];
  }
  int ntiles = Npad >> 4;
  for (int nt=0; nt<ntiles; ++nt){
    v16h b = *(const v16h*)(wp2 + (size_t)nt*512 + lane*16);
    v8f c = {};
    c = __builtin_amdgcn_wmma_f32_16x16x32_f16(false, aH, false, b, (short)0, c,
                                               false, false);
    int n = nt*16 + (lane & 15);
    if (n < N){
      float bias = b2[n];
      #pragma unroll
      for (int r=0; r<8; ++r){
        int m = r + hsel;
        out[(size_t)(e0+m)*N + n] = c[r] + bias;
      }
    }
  }
}

// ---------------------------------------------------------------------------
// Per-edge tensor contraction: out[e,o,p] += sum_{j,i,q} R[e,j,o,i]*B[e,j,p,q]*h[src,i,q]
// ---------------------------------------------------------------------------
__global__ void conv_accum_kernel(const float* __restrict__ R, const float* __restrict__ basisP,
                                  const float* __restrict__ hdeg, const int* __restrict__ src,
                                  float* __restrict__ out,
                                  int nJ, int co, int ci, int pd, int qd, int accumulate, int tot)
{
  int tid = blockIdx.x*blockDim.x + threadIdx.x;
  if (tid >= tot) return;
  int per = co*pd;
  int e = tid / per;
  int op = tid - e*per;
  int o = op / pd, p = op - o*pd;
  const float* Re = R + (size_t)e * (nJ*co*ci);
  const float* Be = basisP + (size_t)e * (nJ*pd*qd);
  const float* h  = hdeg + (size_t)src[e] * (ci*qd);
  float acc = accumulate ? out[(size_t)e*per + op] : 0.f;
  for (int j=0;j<nJ;++j){
    const float* Bj = Be + (j*pd + p)*qd;
    const float* Rj = Re + (j*co + o)*ci;
    for (int i=0;i<ci;++i){
      float t = 0.f;
      for (int q=0;q<qd;++q) t += Bj[q]*h[i*qd+q];
      acc += Rj[i]*t;
    }
  }
  out[(size_t)e*per + op] = acc;
}

// ---------------------------------------------------------------------------
// q = q_mat @ h per node
// ---------------------------------------------------------------------------
__global__ void qproj_kernel(const float* __restrict__ q, const float* __restrict__ h,
                             float* __restrict__ out, int c, int ci, int mdim)
{
  int tid = blockIdx.x*blockDim.x + threadIdx.x;
  int tot = NODES*c*mdim;
  if (tid >= tot) return;
  int n = tid / (c*mdim);
  int om = tid - n*(c*mdim);
  int o = om / mdim, m = om - o*mdim;
  float s = 0.f;
  for (int i=0;i<ci;++i) s += q[o*ci+i]*h[((size_t)n*ci+i)*mdim+m];
  out[tid] = s;
}

// ---------------------------------------------------------------------------
// attention: logits, segment-max (atomic on encoded uint), exp, segment-sum
// ---------------------------------------------------------------------------
__device__ inline unsigned encF(float f){
  unsigned u = __float_as_uint(f);
  return (u & 0x80000000u) ? ~u : (u | 0x80000000u);
}
__device__ inline float decF(unsigned u){
  return __uint_as_float((u & 0x80000000u) ? (u & 0x7FFFFFFFu) : ~u);
}

__global__ void logit_accum_kernel(const float* __restrict__ k, const float* __restrict__ q,
                                   const int* __restrict__ dst, float* __restrict__ logit,
                                   int c, int mdim, int heads)
{
  int tid = blockIdx.x*blockDim.x + threadIdx.x;
  if (tid >= EDGES*heads) return;
  int e = tid / heads, hh = tid - e*heads;
  int ch = c/heads;
  const float* ke = k + (size_t)e*c*mdim + hh*ch*mdim;
  const float* qe = q + (size_t)dst[e]*c*mdim + hh*ch*mdim;
  float s = 0.f;
  for (int i=0;i<ch*mdim;++i) s += ke[i]*qe[i];
  logit[tid] += s;
}

__global__ void logit_max_kernel(const float* __restrict__ logit, const int* __restrict__ dst,
                                 unsigned* __restrict__ menc, float scale, int heads)
{
  int tid = blockIdx.x*blockDim.x + threadIdx.x;
  if (tid >= EDGES*heads) return;
  int e = tid / heads, hh = tid - e*heads;
  atomicMax(&menc[dst[e]*heads+hh], encF(logit[tid]*scale));
}

__global__ void expnum_kernel(const float* __restrict__ logit, const int* __restrict__ dst,
                              const unsigned* __restrict__ menc, float* __restrict__ denom,
                              float* __restrict__ a, float scale, int heads)
{
  int tid = blockIdx.x*blockDim.x + threadIdx.x;
  if (tid >= EDGES*heads) return;
  int e = tid / heads, hh = tid - e*heads;
  float mx = decF(menc[dst[e]*heads+hh]);
  float v = __expf(logit[tid]*scale - mx);
  a[tid] = v;
  atomicAdd(&denom[dst[e]*heads+hh], v);
}

__global__ void zaccum_kernel(const float* __restrict__ v, const float* __restrict__ a,
                              const float* __restrict__ denom, const int* __restrict__ dst,
                              float* __restrict__ z, int cv, int mdim, int heads, int tot)
{
  int tid = blockIdx.x*blockDim.x + threadIdx.x;
  if (tid >= tot) return;
  int per = cv*mdim;
  int e = tid / per;
  int idx = tid - e*per;
  int ch = cv/heads;
  int hh = (idx/mdim)/ch;
  int d = dst[e];
  float w = a[e*heads+hh]/(denom[d*heads+hh]+EPS);
  atomicAdd(&z[(size_t)d*per + idx], w*v[tid]);
}

__global__ void proj_kernel(const float* __restrict__ proj, const float* __restrict__ z,
                            const float* __restrict__ hin, float* __restrict__ out,
                            int cout, int cv, int cinh, int mdim)
{
  int tid = blockIdx.x*blockDim.x + threadIdx.x;
  int tot = NODES*cout*mdim;
  if (tid >= tot) return;
  int n = tid / (cout*mdim);
  int om = tid - n*(cout*mdim);
  int o = om / mdim, m = om - o*mdim;
  int cin = cv + cinh;
  float s = 0.f;
  for (int i=0;i<cv;++i)   s += proj[o*cin+i]   * z[((size_t)n*cv+i)*mdim+m];
  for (int i=0;i<cinh;++i) s += proj[o*cin+cv+i]* hin[((size_t)n*cinh+i)*mdim+m];
  out[tid] = s;
}

__global__ void gnorm_kernel(float* __restrict__ x, const float* __restrict__ b, int c, int mdim)
{
  int tid = blockIdx.x*blockDim.x + threadIdx.x;
  if (tid >= NODES*c) return;
  int cc = tid % c;
  float* xp = x + (size_t)tid*mdim;
  float ss = 0.f;
  for (int m=0;m<mdim;++m) ss += xp[m]*xp[m];
  float nrm = sqrtf(ss + EPS);
  float g = nrm + b[cc];
  g = (g > 0.f ? g : 0.f) / nrm;
  for (int m=0;m<mdim;++m) xp[m] *= g;
}

__global__ void finalize_kernel(const float* __restrict__ h0, const float* __restrict__ h1,
                                float* __restrict__ out)
{
  int tid = blockIdx.x*blockDim.x + threadIdx.x;
  if (tid >= NODES*4) return;
  if (tid < NODES) out[tid] = h0[tid];
  else             out[tid] = h1[tid - NODES];
}

// ---------------------------------------------------------------------------
// Host orchestration
// ---------------------------------------------------------------------------
namespace {

struct Cfg { int nin; int inD[3]; int inC_[3]; int nout; int outD[3]; int outC_[3]; int div_, heads; };
static const Cfg LCFG[3] = {
  {2,{0,1,0},{32,1,0},   3,{0,1,2},{16,16,16}, 4,4},
  {3,{0,1,2},{16,16,16}, 3,{0,1,2},{16,16,16}, 4,4},
  {3,{0,1,2},{16,16,16}, 2,{0,1,0},{1,1,0},    1,1},
};

struct MlpDesc {
  const float *w0,*b0,*w1,*b1,*w2,*b2;
  int N, Npad; size_t packOff; int valid;
};

struct PCur {
  void* const* din; int nin; const float* flat; size_t pos; int leaf;
  const float* take(size_t n){
    if (nin > 7) return (const float*)din[6 + (leaf++)];
    const float* p = flat + pos; pos += n; return p;
  }
};

} // namespace

extern "C" void kernel_launch(void* const* d_in, const int* in_sizes, int n_in,
                              void* d_out, int out_size, void* d_ws, size_t ws_size,
                              hipStream_t stream)
{
  (void)in_sizes; (void)out_size; (void)ws_size;
  const float* node_pos = (const float*)d_in[0];
  const int*   edge_src = (const int*)d_in[1];
  const int*   edge_dst = (const int*)d_in[2];
  const float* edge_feat= (const float*)d_in[3];
  const float* t0       = (const float*)d_in[4];
  const float* t1       = (const float*)d_in[5];

  // ---------------- param walk (jax pytree flatten order: sorted keys) ------
  PCur pc{d_in, n_in, (const float*)d_in[6], 0, 0};
  static MlpDesc VR[3][3][3], KR[3][3][3];        // [layer][di][do]
  const float *Qp[3][3]={}, *PRp[3][3]={}, *NBp[3][3]={};
  for (int li=0; li<3; ++li)
    for (int a=0;a<3;++a) for (int b=0;b<3;++b){ VR[li][a][b].valid=0; KR[li][a][b].valid=0; }

  size_t packHalves = 0;
  for (int li=0; li<3; ++li){
    const Cfg& c = LCFG[li];
    bool isIn[3]={false,false,false}, isOut[3]={false,false,false};
    int inC[3]={0,0,0}, outC[3]={0,0,0}, valC[3]={0,0,0};
    for (int i=0;i<c.nin;++i){ isIn[c.inD[i]]=true; inC[c.inD[i]]=c.inC_[i]; }
    for (int i=0;i<c.nout;++i){ isOut[c.outD[i]]=true; outC[c.outD[i]]=c.outC_[i]; }
    bool isKey[3];
    for (int d=0;d<3;++d){
      valC[d] = isOut[d] ? (outC[d]/c.div_ > 0 ? outC[d]/c.div_ : 1) : 0;
      isKey[d] = isOut[d] && isIn[d];
    }
    auto walkRadial = [&](MlpDesc& m, int di, int dg){
      int mn = di<dg?di:dg;
      int outDim = (2*mn+1) * valC[dg] * inC[di];
      m.b0 = pc.take(32); m.b1 = pc.take(32); m.b2 = pc.take(outDim);
      m.w0 = pc.take(33*32); m.w1 = pc.take(32*32); m.w2 = pc.take(32*(size_t)outDim);
      m.N = outDim; m.Npad = (outDim+15)&~15;
      m.packOff = packHalves;
      packHalves += (size_t)(6 + m.Npad/16) * 512;
      m.valid = 1;
    };
    // 'k'
    for (int di=0; di<3; ++di) if (isIn[di])
      for (int dg=0; dg<3; ++dg) if (isKey[dg]) walkRadial(KR[li][di][dg], di, dg);
    // 'nb%d'
    for (int dg=0; dg<3; ++dg) if (isOut[dg]) NBp[li][dg] = pc.take(outC[dg]);
    // 'proj%d'
    for (int dg=0; dg<3; ++dg) if (isOut[dg]){
      int cin = valC[dg] + (isIn[dg]?inC[dg]:0);
      PRp[li][dg] = pc.take((size_t)outC[dg]*cin);
    }
    // 'q%d'
    for (int dg=0; dg<3; ++dg) if (isKey[dg]) Qp[li][dg] = pc.take((size_t)valC[dg]*inC[dg]);
    // 'v'
    for (int di=0; di<3; ++di) if (isIn[di])
      for (int dg=0; dg<3; ++dg) if (isOut[dg]) walkRadial(VR[li][di][dg], di, dg);
  }

  // ---------------- workspace carve ----------------------------------------
  char* wsb = (char*)d_ws;
  size_t cur = 0;
  auto walloc = [&](size_t bytes)->void*{
    void* p = wsb + cur;
    cur = (cur + bytes + 255) & ~(size_t)255;
    return p;
  };
  float*    wsRcg  = (float*)walloc(1225*sizeof(float));
  float*    wsY    = (float*)walloc((size_t)EDGES*25*sizeof(float));
  _Float16* wsRf   = (_Float16*)walloc((size_t)EDGES*64*sizeof(_Float16));
  float*    wsBasis= (float*)walloc((size_t)EDGES*259*sizeof(float));
  float*    wsR    = (float*)walloc((size_t)EDGES*320*sizeof(float));
  _Float16* wpack  = (_Float16*)walloc(packHalves*sizeof(_Float16));
  float* hbuf[2][3];
  for (int s=0;s<2;++s) for (int d=0;d<3;++d) hbuf[s][d] = (float*)walloc((size_t)NODES*80*sizeof(float));
  float* vout[3]; float* kout[3]; float* qbuf[3]; float* zbuf[3];
  for (int d=0;d<3;++d){
    vout[d] = (float*)walloc((size_t)EDGES*20*sizeof(float));
    kout[d] = (float*)walloc((size_t)EDGES*20*sizeof(float));
    qbuf[d] = (float*)walloc((size_t)NODES*20*sizeof(float));
    zbuf[d] = (float*)walloc((size_t)NODES*20*sizeof(float));
  }
  float*    logitB = (float*)walloc((size_t)EDGES*4*sizeof(float));
  float*    aB     = (float*)walloc((size_t)EDGES*4*sizeof(float));
  unsigned* menc   = (unsigned*)walloc((size_t)NODES*4*sizeof(unsigned));
  float*    denom  = (float*)walloc((size_t)NODES*4*sizeof(float));

  // basis pair offsets (floats)
  size_t pairOff[9];
  { size_t o = 0;
    for (int di=0; di<3; ++di)
      for (int dg=0; dg<3; ++dg){
        pairOff[di*3+dg] = o;
        int mn = di<dg?di:dg;
        o += (size_t)EDGES * (2*mn+1)*(2*dg+1)*(2*di+1);
      }
  }

  // ---------------- launches ------------------------------------------------
  rcg_kernel<<<1,32,0,stream>>>(wsRcg);
  edge_geom_kernel<<<EDGES/256,256,0,stream>>>(node_pos, edge_src, edge_dst, edge_feat, wsRf, wsY);
  basis_kernel<<<EDGES/64,64,0,stream>>>(wsY, wsRcg, wsBasis);

  for (int li=0; li<3; ++li)
    for (int di=0; di<3; ++di)
      for (int dg=0; dg<3; ++dg){
        if (VR[li][di][dg].valid){
          const MlpDesc& m = VR[li][di][dg];
          pack_mlp_kernel<<<6 + m.Npad/16, 32, 0, stream>>>(m.w0, m.w1, m.w2, wpack+m.packOff, m.N, m.Npad);
        }
        if (KR[li][di][dg].valid){
          const MlpDesc& m = KR[li][di][dg];
          pack_mlp_kernel<<<6 + m.Npad/16, 32, 0, stream>>>(m.w0, m.w1, m.w2, wpack+m.packOff, m.N, m.Npad);
        }
      }

  const float* hin[3] = {t0, t1, nullptr};
  for (int li=0; li<3; ++li){
    const Cfg& c = LCFG[li];
    bool isIn[3]={false,false,false}, isOut[3]={false,false,false};
    int inC[3]={0,0,0}, outC[3]={0,0,0}, valC[3]={0,0,0};
    for (int i=0;i<c.nin;++i){ isIn[c.inD[i]]=true; inC[c.inD[i]]=c.inC_[i]; }
    for (int i=0;i<c.nout;++i){ isOut[c.outD[i]]=true; outC[c.outD[i]]=c.outC_[i]; }
    bool isKey[3];
    for (int d=0;d<3;++d){
      valC[d] = isOut[d] ? (outC[d]/c.div_ > 0 ? outC[d]/c.div_ : 1) : 0;
      isKey[d] = isOut[d] && isIn[d];
    }
    int heads = c.heads;
    float* hout[3] = { hbuf[li&1][0], hbuf[li&1][1], hbuf[li&1][2] };

    // value & key convolutions: WMMA MLP -> per-edge CG contraction
    for (int br=0; br<2; ++br){
      for (int dg=0; dg<3; ++dg){
        bool used = br==0 ? isOut[dg] : isKey[dg];
        if (!used) continue;
        int first = 1;
        for (int di=0; di<3; ++di){
          if (!isIn[di]) continue;
          const MlpDesc& m = br ? KR[li][di][dg] : VR[li][di][dg];
          mlp_wmma_kernel<<<EDGES/16,32,0,stream>>>(wsRf, wpack+m.packOff,
                                                    m.b0, m.b1, m.b2, wsR, m.N, m.Npad);
          int mn = di<dg?di:dg;
          int nJ = 2*mn+1, pd = 2*dg+1, qd = 2*di+1, co = valC[dg];
          int tot = EDGES*co*pd;
          conv_accum_kernel<<<CDIV(tot,256),256,0,stream>>>(
              wsR, wsBasis + pairOff[di*3+dg], hin[di], edge_src,
              (br ? kout : vout)[dg], nJ, co, inC[di], pd, qd, first?0:1, tot);
          first = 0;
        }
      }
    }

    // attention
    int dk = 0;
    for (int dg=0; dg<3; ++dg) if (isKey[dg]) dk += (valC[dg]/heads)*(2*dg+1);
    float scale = 1.0f/sqrtf((float)dk);

    for (int dg=0; dg<3; ++dg) if (isKey[dg]){
      int tot = NODES*valC[dg]*(2*dg+1);
      qproj_kernel<<<CDIV(tot,256),256,0,stream>>>(Qp[li][dg], hin[dg], qbuf[dg],
                                                   valC[dg], inC[dg], 2*dg+1);
    }
    hipMemsetAsync(logitB, 0, (size_t)EDGES*heads*sizeof(float), stream);
    for (int dg=0; dg<3; ++dg) if (isKey[dg])
      logit_accum_kernel<<<CDIV(EDGES*heads,256),256,0,stream>>>(
          kout[dg], qbuf[dg], edge_dst, logitB, valC[dg], 2*dg+1, heads);
    hipMemsetAsync(menc,  0, (size_t)NODES*heads*sizeof(unsigned), stream);
    hipMemsetAsync(denom, 0, (size_t)NODES*heads*sizeof(float), stream);
    logit_max_kernel<<<CDIV(EDGES*heads,256),256,0,stream>>>(logitB, edge_dst, menc, scale, heads);
    expnum_kernel<<<CDIV(EDGES*heads,256),256,0,stream>>>(logitB, edge_dst, menc, denom, aB, scale, heads);

    for (int dg=0; dg<3; ++dg) if (isOut[dg]){
      int per = valC[dg]*(2*dg+1);
      hipMemsetAsync(zbuf[dg], 0, (size_t)NODES*per*sizeof(float), stream);
      int tot = EDGES*per;
      zaccum_kernel<<<CDIV(tot,256),256,0,stream>>>(vout[dg], aB, denom, edge_dst,
                                                    zbuf[dg], valC[dg], 2*dg+1, heads, tot);
    }
    for (int dg=0; dg<3; ++dg) if (isOut[dg]){
      int tot = NODES*outC[dg]*(2*dg+1);
      proj_kernel<<<CDIV(tot,256),256,0,stream>>>(PRp[li][dg], zbuf[dg],
                                                  isIn[dg]?hin[dg]:nullptr, hout[dg],
                                                  outC[dg], valC[dg],
                                                  isIn[dg]?inC[dg]:0, 2*dg+1);
    }
    if (li < 2)
      for (int dg=0; dg<3; ++dg) if (isOut[dg])
        gnorm_kernel<<<CDIV(NODES*outC[dg],256),256,0,stream>>>(hout[dg], NBp[li][dg],
                                                                outC[dg], 2*dg+1);
    for (int dg=0; dg<3; ++dg) hin[dg] = isOut[dg] ? hout[dg] : nullptr;
  }

  finalize_kernel<<<CDIV(NODES*4,256),256,0,stream>>>(hin[0], hin[1], (float*)d_out);
}
